// ProbAttention_4011499454602
// MI455X (gfx1250) — matrix operand order repaired and checked by reading, not compiled, not run on hardware
//
#include <hip/hip_runtime.h>
#include <math.h>

// Problem constants (match reference)
#define B_  4
#define L_  2048
#define H_  8
#define D_  64
#define S_  2048
#define U_  40
#define EPS_ 1e-8f
#define NORM_EPS_ 1e-12f

typedef __attribute__((ext_vector_type(16))) _Float16 v16h;
typedef __attribute__((ext_vector_type(8)))  float    v8f;

// ---------- wave reductions (wave32) ----------
__device__ __forceinline__ float wsum32(float x) {
#pragma unroll
  for (int m = 1; m < 32; m <<= 1) x += __shfl_xor(x, m, 32);
  return x;
}
// reductions within each 16-lane half (masks < 16 never cross halves)
__device__ __forceinline__ float rowmax16(float x) {
#pragma unroll
  for (int m = 1; m < 16; m <<= 1) x = fmaxf(x, __shfl_xor(x, m, 32));
  return x;
}
__device__ __forceinline__ float rowsum16(float x) {
#pragma unroll
  for (int m = 1; m < 16; m <<= 1) x += __shfl_xor(x, m, 32);
  return x;
}

// ---------- kernel 0: zero the output ----------
__global__ void k_zero(float* __restrict__ p, long n) {
  long i = (long)blockIdx.x * blockDim.x + threadIdx.x;
  long stride = (long)gridDim.x * blockDim.x;
  for (; i < n; i += stride) p[i] = 0.0f;
}

// ---------- kernel 1: sampled NPCC score M[b,h,l] ----------
// One wave per query. Lane j holds dims j and j+32.
__global__ __launch_bounds__(256)
void k_sample_scores(const float* __restrict__ q, const float* __restrict__ k,
                     const int* __restrict__ idx, float* __restrict__ M) {
  const int lane = threadIdx.x & 31;
  const long gq = (long)blockIdx.x * 8 + (threadIdx.x >> 5);  // (b*H+h)*L + l
  const int l  = (int)(gq % L_);
  const int bh = (int)(gq / L_);
  const int h  = bh % H_;
  const int b  = bh / H_;

  const float* qrow = q + (((long)b * L_ + l) * H_ + h) * D_;
  float q0 = qrow[lane], q1 = qrow[lane + 32];
  const float mean = wsum32(q0 + q1) * (1.0f / 64.0f);
  q0 -= mean; q1 -= mean;
  const float nq = sqrtf(wsum32(q0 * q0 + q1 * q1));

  const int* ip = idx + gq * U_;
  float acc = 0.0f;
  for (int u = 0; u < U_; ++u) {
    const int s = ip[u];
    const float* krow = k + (((long)b * S_ + s) * H_ + h) * D_;
    float k0 = krow[lane], k1 = krow[lane + 32];
    const float km = wsum32(k0 + k1) * (1.0f / 64.0f);
    k0 -= km; k1 -= km;
    const float cov = wsum32(q0 * k0 + q1 * k1);
    const float nk  = sqrtf(wsum32(k0 * k0 + k1 * k1));
    acc += fabsf(cov / (nq * nk + EPS_));
  }
  if (lane == 0) M[gq] = acc * (1.0f / (float)U_);
}

// ---------- kernel 2: top-U selection per (b,h) ----------
__global__ __launch_bounds__(256)
void k_topk(const float* __restrict__ M, int* __restrict__ M_top) {
  __shared__ float vals[L_];
  __shared__ float rv[256];
  __shared__ int   ri[256];
  const int bh = blockIdx.x, tid = threadIdx.x;
  for (int i = tid; i < L_; i += 256) vals[i] = M[(long)bh * L_ + i];
  __syncthreads();
  for (int it = 0; it < U_; ++it) {
    float bv = -INFINITY; int bi = L_;
    for (int i = tid; i < L_; i += 256) {
      const float x = vals[i];
      if (x > bv || (x == bv && i < bi)) { bv = x; bi = i; }
    }
    rv[tid] = bv; ri[tid] = bi;
    __syncthreads();
    for (int off = 128; off > 0; off >>= 1) {
      if (tid < off) {
        const float ov = rv[tid + off]; const int oi = ri[tid + off];
        if (ov > rv[tid] || (ov == rv[tid] && oi < ri[tid])) { rv[tid] = ov; ri[tid] = oi; }
      }
      __syncthreads();
    }
    if (tid == 0) { M_top[bh * U_ + it] = ri[0]; vals[ri[0]] = -INFINITY; }
    __syncthreads();
  }
}

// ---------- kernel 3: flash attention over the selected queries (WMMA) ----------
// One wave per (qtile, h, b); qtile covers 16 query rows (40 padded to 48).
__global__ __launch_bounds__(32)
void k_attn(const float* __restrict__ q, const float* __restrict__ k,
            const float* __restrict__ v, const int* __restrict__ M_top,
            float* __restrict__ out) {
  __shared__ _Float16 Qsh[16][D_];
  __shared__ _Float16 Ksh[32][D_ + 8];
  __shared__ _Float16 Vsh[32][D_ + 8];
  __shared__ _Float16 Psh[16][32];

  const int qtile = blockIdx.x;   // 0..2
  const int h     = blockIdx.y;
  const int b     = blockIdx.z;
  const int lane  = threadIdx.x;
  const int half  = lane >> 4;    // 0 or 1
  const int lm    = lane & 15;

  // ---- load + L2-normalize 16 Q rows into LDS as f16 ----
  for (int r = 0; r < 16; ++r) {
    const int u = qtile * 16 + r;          // wave-uniform
    float x0 = 0.0f, x1 = 0.0f;
    if (u < U_) {
      const int l = M_top[(b * H_ + h) * U_ + u];
      const float* qr = q + (((long)b * L_ + l) * H_ + h) * D_;
      x0 = qr[lane]; x1 = qr[lane + 32];
      const float inv = 1.0f / fmaxf(sqrtf(wsum32(x0 * x0 + x1 * x1)), NORM_EPS_);
      x0 *= inv; x1 *= inv;
    }
    Qsh[r][lane] = (_Float16)x0;
    Qsh[r][lane + 32] = (_Float16)x1;
  }
  __syncthreads();

  // ---- build A fragments (16x32 f16) for the two 32-dim chunks ----
  v16h a0, a1;
#pragma unroll
  for (int e = 0; e < 16; ++e) {
    const int vg = e >> 1;
    const int dloc = half * 16 + e;
    a0[e] = Qsh[lm][dloc];
    a1[e] = Qsh[lm][32 + dloc];
  }

  v8f o0 = {}, o1 = {}, o2 = {}, o3 = {};
  float m_run[8], l_run[8];
#pragma unroll
  for (int r = 0; r < 8; ++r) { m_run[r] = -1e30f; l_run[r] = 0.0f; }
  const float log_scale = (1.0f / 0.24f) * logf((float)S_);

  for (int s0 = 0; s0 < S_; s0 += 32) {
    // ---- stage 32 K rows (L2-normalized) and 32 V rows as f16 ----
    {
      const int s = s0 + lane;
      const float* kr = k + (((long)b * S_ + s) * H_ + h) * D_;
      const float* vr = v + (((long)b * S_ + s) * H_ + h) * D_;
      float nsq = 0.0f;
#pragma unroll
      for (int d = 0; d < D_; ++d) { const float x = kr[d]; nsq += x * x; }
      const float inv = 1.0f / fmaxf(sqrtf(nsq), NORM_EPS_);
#pragma unroll
      for (int d = 0; d < D_; ++d) Ksh[lane][d] = (_Float16)(kr[d] * inv);
#pragma unroll
      for (int d = 0; d < D_; ++d) Vsh[lane][d] = (_Float16)vr[d];
    }
    __syncthreads();

    // ---- scores: c[sub] = (Q tile) @ (K sub-tile)^T * log_scale ----
    v8f c[2];
#pragma unroll
    for (int sub = 0; sub < 2; ++sub) {
      const int key = sub * 16 + lm;       // B-operand column (N)
      v16h b0, b1;
#pragma unroll
      for (int e = 0; e < 16; ++e) {
        b0[e] = Ksh[key][half * 16 + e];          // dims 0..31
        b1[e] = Ksh[key][32 + half * 16 + e];     // dims 32..63
      }
      v8f acc = {};
      acc = __builtin_amdgcn_wmma_f32_16x16x32_f16(false, a0, false, b0,
                                                   (short)0, acc, false, false);
      acc = __builtin_amdgcn_wmma_f32_16x16x32_f16(false, a1, false, b1,
                                                   (short)0, acc, false, false);
      c[sub] = acc * log_scale;
    }

    // ---- online softmax update (row = half*8 + r) ----
    float alpha[8];
#pragma unroll
    for (int r = 0; r < 8; ++r) {
      float mx = rowmax16(fmaxf(c[0][r], c[1][r]));
      const float mnew = fmaxf(m_run[r], mx);
      alpha[r] = __expf(m_run[r] - mnew);
      m_run[r] = mnew;
      const float p0 = __expf(c[0][r] - mnew);
      const float p1 = __expf(c[1][r] - mnew);
      l_run[r] = l_run[r] * alpha[r] + rowsum16(p0 + p1);
      Psh[half * 8 + r][lm]      = (_Float16)p0;
      Psh[half * 8 + r][16 + lm] = (_Float16)p1;
      o0[r] *= alpha[r]; o1[r] *= alpha[r]; o2[r] *= alpha[r]; o3[r] *= alpha[r];
    }
    __syncthreads();

    // ---- O += P @ V  (A = P 16x32 f16, B = V 32x16 per 16-dim tile) ----
    v16h pA;
#pragma unroll
    for (int e = 0; e < 16; ++e) {
      const int vg = e >> 1;
      const int kk = half * 16 + e;
      pA[e] = Psh[lm][kk];
    }
    {
      v16h bv;
#pragma unroll
      for (int e = 0; e < 16; ++e) bv[e] = Vsh[half * 16 + e][0 * 16 + lm];
      o0 = __builtin_amdgcn_wmma_f32_16x16x32_f16(false, pA, false, bv, (short)0, o0, false, false);
#pragma unroll
      for (int e = 0; e < 16; ++e) bv[e] = Vsh[half * 16 + e][1 * 16 + lm];
      o1 = __builtin_amdgcn_wmma_f32_16x16x32_f16(false, pA, false, bv, (short)0, o1, false, false);
#pragma unroll
      for (int e = 0; e < 16; ++e) bv[e] = Vsh[half * 16 + e][2 * 16 + lm];
      o2 = __builtin_amdgcn_wmma_f32_16x16x32_f16(false, pA, false, bv, (short)0, o2, false, false);
#pragma unroll
      for (int e = 0; e < 16; ++e) bv[e] = Vsh[half * 16 + e][3 * 16 + lm];
      o3 = __builtin_amdgcn_wmma_f32_16x16x32_f16(false, pA, false, bv, (short)0, o3, false, false);
    }
    __syncthreads();
  }

  // ---- epilogue: scatter normalized rows to out[b, l, h, :] ----
#pragma unroll
  for (int r = 0; r < 8; ++r) {
    const int u = qtile * 16 + half * 8 + r;
    if (u < U_) {
      const int l = M_top[(b * H_ + h) * U_ + u];
      float* orow = out + (((long)b * L_ + l) * H_ + h) * D_;
      const float invl = 1.0f / l_run[r];
      orow[0 * 16 + lm] = o0[r] * invl;
      orow[1 * 16 + lm] = o1[r] * invl;
      orow[2 * 16 + lm] = o2[r] * invl;
      orow[3 * 16 + lm] = o3[r] * invl;
    }
  }
}

extern "C" void kernel_launch(void* const* d_in, const int* in_sizes, int n_in,
                              void* d_out, int out_size, void* d_ws, size_t ws_size,
                              hipStream_t stream) {
  const float* q   = (const float*)d_in[0];
  const float* k   = (const float*)d_in[1];
  const float* v   = (const float*)d_in[2];
  const int*   idx = (const int*)d_in[3];
  float* out = (float*)d_out;

  float* Mbuf = (float*)d_ws;                                    // B*H*L floats
  int*   Mtop = (int*)((char*)d_ws + (size_t)B_ * H_ * L_ * sizeof(float));

  k_zero<<<1024, 256, 0, stream>>>(out, (long)out_size);
  k_sample_scores<<<(B_ * H_ * L_) / 8, 256, 0, stream>>>(q, k, idx, Mbuf);
  k_topk<<<B_ * H_, 256, 0, stream>>>(Mbuf, Mtop);
  k_attn<<<dim3((U_ + 15) / 16, H_, B_), 32, 0, stream>>>(q, k, v, Mtop, out);
}
